// AsrModel_16252156248538
// MI455X (gfx1250) — compile-verified, matrix operand order checked by reading, not run on hardware
//
#include <hip/hip_runtime.h>
#include <hip/hip_bf16.h>

// Problem dims (B, T, F, H, D, L) = (64, 1024, 304, 304, 128, 128)
#define B_    64
#define T_    1024
#define F_    304
#define H_    304
#define G3_   912      // 3*H gate rows
#define D_    128      // vocab
#define L_    128      // decode steps
#define KP_   320      // K padded to 10 * 32 (WMMA K)
#define GP_   1024     // gate rows padded to 16 WGs * 64
#define NWG_  16
#define NTHR_ 256      // 8 wave32 waves per WG

typedef _Float16 half8 __attribute__((ext_vector_type(8)));
typedef _Float16 v16h  __attribute__((ext_vector_type(16)));
typedef float    v8f   __attribute__((ext_vector_type(8)));

// Load one 16-lane-slice WMMA fragment: two contiguous 16B half8 chunks
// per the CDNA5 16-bit A/B VGPR layout (lane holds K {kb..kb+7} and {kb+16..kb+23}).
__device__ __forceinline__ v16h ld_frag(const _Float16* p) {
    half8 lo = *(const half8*)(p);
    half8 hh = *(const half8*)(p + 16);
    v16h r;
#pragma unroll
    for (int i = 0; i < 8; ++i) { r[i] = lo[i]; r[i + 8] = hh[i]; }
    return r;
}

// CDNA5 async DMA: 16 bytes global -> LDS, tracked by ASYNCcnt (no VGPR data).
// GVS mode: lds_addr VGPR, 32-bit per-lane byte offset VGPR, 64-bit SGPR base.
__device__ __forceinline__ void async_copy_b128(unsigned lds_byte_addr,
                                                const void* gbase, int byte_off) {
    asm volatile("global_load_async_to_lds_b128 %0, %1, %2 offset:0"
                 :: "v"(lds_byte_addr), "v"(byte_off), "s"(gbase)
                 : "memory");
}
__device__ __forceinline__ void wait_async0() {
    asm volatile("s_wait_asynccnt 0" ::: "memory");
}
__device__ __forceinline__ unsigned lds_addr_of(const void* p) {
    return (unsigned)(size_t)p;   // LDS aperture: addr[31:0] is the LDS byte offset
}

// Device-wide arrive/phase barrier across the NWG_ resident workgroups.
__device__ __forceinline__ void gbarrier(int* cnt, int* phase, int* lphase) {
    __threadfence();
    __syncthreads();
    if (threadIdx.x == 0) {
        int target = *lphase + 1;
        if (atomicAdd(cnt, 1) == NWG_ - 1) {
            __hip_atomic_store(cnt, 0, __ATOMIC_RELAXED, __HIP_MEMORY_SCOPE_AGENT);
            atomicAdd(phase, 1);
        } else {
            while (__hip_atomic_load(phase, __ATOMIC_RELAXED, __HIP_MEMORY_SCOPE_AGENT) < target)
                __builtin_amdgcn_s_sleep(2);
        }
        *lphase = target;
    }
    __syncthreads();
    __threadfence();
}

// f32 -> padded f16 weight tiles: dst is rowsPad x KP_, zero outside src.
__global__ void pad_convert(const float* __restrict__ src, _Float16* __restrict__ dst,
                            int rowsPad, int srcRows, int srcCols) {
    int total = rowsPad * KP_;
    for (int i = blockIdx.x * blockDim.x + threadIdx.x; i < total;
         i += gridDim.x * blockDim.x) {
        int r = i / KP_, k = i - r * KP_;
        float v = (r < srcRows && k < srcCols) ? src[r * srcCols + k] : 0.f;
        dst[i] = (_Float16)v;
    }
}

__global__ void copy_target(const int* __restrict__ target, float* __restrict__ out) {
    int i = blockIdx.x * blockDim.x + threadIdx.x;
    if (i < B_ * (L_ - 1)) {
        int b = i / (L_ - 1), l = i - b * (L_ - 1) + 1;
        out[i] = (float)target[b * L_ + l];
    }
}

// ---------------- Encoder: 1024-step GRU, weights resident in LDS -------------
__global__ void __launch_bounds__(NTHR_) enc_gru(
    const float* __restrict__ x, const float* __restrict__ bih, const float* __restrict__ bhh,
    const _Float16* __restrict__ Wih_h, const _Float16* __restrict__ Whh_h,
    float* __restrict__ hf32, _Float16* __restrict__ hf16,
    float* __restrict__ Gi, float* __restrict__ Gh, int* ctrl)
{
    extern __shared__ _Float16 smem[];
    _Float16* Ax = smem;                 // x_t fragment matrix  (64 x KP_)
    _Float16* Ah = Ax + 64 * KP_;        // h   fragment matrix  (64 x KP_)
    _Float16* W0 = Ah + 64 * KP_;        // Wih slice (64 gate rows x KP_)
    _Float16* W1 = W0 + 64 * KP_;        // Whh slice

    const int tid  = threadIdx.x;
    const int wg   = blockIdx.x;
    const int wid  = tid >> 5;
    const int lane = tid & 31;
    const int row  = lane & 15;
    const int hih  = lane >> 4;

    int* cnt = ctrl;
    int* phase = ctrl + 1;
    int lphase = 0;
    if (tid == 0)
        lphase = __hip_atomic_load(phase, __ATOMIC_RELAXED, __HIP_MEMORY_SCOPE_AGENT);

    const unsigned axB = lds_addr_of(Ax);
    const unsigned ahB = lds_addr_of(Ah);
    const unsigned w0B = lds_addr_of(W0);
    const unsigned w1B = lds_addr_of(W1);

    // Pin this WG's weight slice in LDS once via async DMA (2 x 40 KB).
    for (int i = tid * 8; i < 64 * KP_; i += NTHR_ * 8) {
        async_copy_b128(w0B + i * 2, Wih_h, (wg * 64 * KP_ + i) * 2);
        async_copy_b128(w1B + i * 2, Whh_h, (wg * 64 * KP_ + i) * 2);
    }

    for (int t = 0; t < T_; ++t) {
        // Async DMA the f16 h state into LDS (overlaps the x_t convert below).
        for (int i = tid * 8; i < 64 * KP_; i += NTHR_ * 8)
            async_copy_b128(ahB + i * 2, hf16, i * 2);

        // Stage x_t (f32->f16 + zero pad) through VALU.
        for (int i = tid; i < 64 * KP_; i += NTHR_) {
            int b = i / KP_, k = i - b * KP_;
            float xv = (k < F_) ? x[(b * T_ + t) * F_ + k] : 0.f;
            Ax[i] = (_Float16)xv;
            if ((k < F_) && ((k & 31) == 0) && (t + 1 < T_))
                __builtin_prefetch(&x[(b * T_ + t + 1) * F_ + k], 0, 1);
        }
        (void)axB;
        wait_async0();
        __syncthreads();

        // Two 64x64 GEMMs (Gi = x·Wih^T, Gh = h·Whh^T): 32 C-tiles over 8 waves.
#pragma unroll
        for (int tt = 0; tt < 4; ++tt) {
            int idx = wid + tt * 8;                 // 0..31
            const _Float16* Ab = (idx < 16) ? Ax : Ah;
            const _Float16* Wb = (idx < 16) ? W0 : W1;
            int t16 = idx & 15;
            int mt = t16 >> 2, nt = t16 & 3;
            v8f c = {};
#pragma unroll
            for (int k0 = 0; k0 < KP_; k0 += 32) {
                v16h a  = ld_frag(Ab + (mt * 16 + row) * KP_ + k0 + hih * 8);
                v16h bm = ld_frag(Wb + (nt * 16 + row) * KP_ + k0 + hih * 8);
                c = __builtin_amdgcn_wmma_f32_16x16x32_f16(
                        false, a, false, bm, (short)0, c, false, false);
            }
            float* Gd = (idx < 16) ? Gi : Gh;
            int gcol = wg * 64 + nt * 16 + row;
#pragma unroll
            for (int r = 0; r < 8; ++r)
                Gd[(mt * 16 + r + hih * 8) * GP_ + gcol] = c[r];
        }

        gbarrier(cnt, phase, &lphase);

        // Elementwise GRU gate combine, partitioned across all 16 WGs.
        for (int i = wg * NTHR_ + tid; i < B_ * H_; i += NWG_ * NTHR_) {
            int b = i / H_, j = i - b * H_;
            float ir  = Gi[b * GP_ + j]          + bih[j];
            float hr  = Gh[b * GP_ + j]          + bhh[j];
            float iz  = Gi[b * GP_ + H_ + j]     + bih[H_ + j];
            float hz  = Gh[b * GP_ + H_ + j]     + bhh[H_ + j];
            float in_ = Gi[b * GP_ + 2 * H_ + j] + bih[2 * H_ + j];
            float hn  = Gh[b * GP_ + 2 * H_ + j] + bhh[2 * H_ + j];
            float rr = 1.f / (1.f + __expf(-(ir + hr)));
            float zz = 1.f / (1.f + __expf(-(iz + hz)));
            float nn = tanhf(in_ + rr * hn);
            float ho = hf32[i];
            float hv = (1.f - zz) * nn + zz * ho;
            hf32[i] = hv;
            hf16[b * KP_ + j] = (_Float16)hv;
        }

        gbarrier(cnt, phase, &lphase);
    }
}

// ---------------- Decoder: 128-step greedy GRU + WMMA logits -----------------
__global__ void __launch_bounds__(NTHR_) dec_gru(
    const int* __restrict__ target, const _Float16* __restrict__ emb_h,
    const float* __restrict__ bih, const float* __restrict__ bhh,
    const _Float16* __restrict__ Wih_h, const _Float16* __restrict__ Whh_h,
    const _Float16* __restrict__ lin_h, const float* __restrict__ lin_b,
    float* __restrict__ hf32, _Float16* __restrict__ hf16,
    float* __restrict__ Gi, float* __restrict__ Gh,
    float* __restrict__ logits, int* __restrict__ tok,
    float* __restrict__ out_soft, float* __restrict__ out_asr, int* ctrl)
{
    extern __shared__ _Float16 smem[];
    _Float16* Ae = smem;
    _Float16* Ah = Ae + 64 * KP_;
    _Float16* W0 = Ah + 64 * KP_;
    _Float16* W1 = W0 + 64 * KP_;
    __shared__ int stok[B_];

    const int tid  = threadIdx.x;
    const int wg   = blockIdx.x;
    const int wid  = tid >> 5;
    const int lane = tid & 31;
    const int row  = lane & 15;
    const int hih  = lane >> 4;

    int* cnt = ctrl;
    int* phase = ctrl + 1;
    int lphase = 0;
    if (tid == 0)
        lphase = __hip_atomic_load(phase, __ATOMIC_RELAXED, __HIP_MEMORY_SCOPE_AGENT);

    const unsigned aeB = lds_addr_of(Ae);
    const unsigned ahB = lds_addr_of(Ah);
    const unsigned w0B = lds_addr_of(W0);
    const unsigned w1B = lds_addr_of(W1);

    if (wg == 0 && tid < B_) tok[tid] = target[tid * L_];  // tok0 = target[:,0,0]
    for (int i = tid * 8; i < 64 * KP_; i += NTHR_ * 8) {
        async_copy_b128(w0B + i * 2, Wih_h, (wg * 64 * KP_ + i) * 2);
        async_copy_b128(w1B + i * 2, Whh_h, (wg * 64 * KP_ + i) * 2);
    }
    wait_async0();
    gbarrier(cnt, phase, &lphase);

    for (int s = 0; s < L_; ++s) {
        if (tid < B_) stok[tid] = tok[tid];
        __syncthreads();
        // Async DMA: gather embedded token rows + copy h, both pure f16 moves.
        for (int i = tid * 8; i < 64 * KP_; i += NTHR_ * 8) {
            int b = i / KP_, k = i - b * KP_;
            async_copy_b128(aeB + i * 2, emb_h, (stok[b] * KP_ + k) * 2);
            async_copy_b128(ahB + i * 2, hf16, i * 2);
        }
        wait_async0();
        __syncthreads();

#pragma unroll
        for (int tt = 0; tt < 4; ++tt) {
            int idx = wid + tt * 8;
            const _Float16* Ab = (idx < 16) ? Ae : Ah;
            const _Float16* Wb = (idx < 16) ? W0 : W1;
            int t16 = idx & 15;
            int mt = t16 >> 2, nt = t16 & 3;
            v8f c = {};
#pragma unroll
            for (int k0 = 0; k0 < KP_; k0 += 32) {
                v16h a  = ld_frag(Ab + (mt * 16 + row) * KP_ + k0 + hih * 8);
                v16h bm = ld_frag(Wb + (nt * 16 + row) * KP_ + k0 + hih * 8);
                c = __builtin_amdgcn_wmma_f32_16x16x32_f16(
                        false, a, false, bm, (short)0, c, false, false);
            }
            float* Gd = (idx < 16) ? Gi : Gh;
            int gcol = wg * 64 + nt * 16 + row;
#pragma unroll
            for (int r = 0; r < 8; ++r)
                Gd[(mt * 16 + r + hih * 8) * GP_ + gcol] = c[r];
        }

        gbarrier(cnt, phase, &lphase);

        for (int i = wg * NTHR_ + tid; i < B_ * H_; i += NWG_ * NTHR_) {
            int b = i / H_, j = i - b * H_;
            float ir  = Gi[b * GP_ + j]          + bih[j];
            float hr  = Gh[b * GP_ + j]          + bhh[j];
            float iz  = Gi[b * GP_ + H_ + j]     + bih[H_ + j];
            float hz  = Gh[b * GP_ + H_ + j]     + bhh[H_ + j];
            float in_ = Gi[b * GP_ + 2 * H_ + j] + bih[2 * H_ + j];
            float hn  = Gh[b * GP_ + 2 * H_ + j] + bhh[2 * H_ + j];
            float rr = 1.f / (1.f + __expf(-(ir + hr)));
            float zz = 1.f / (1.f + __expf(-(iz + hz)));
            float nn = tanhf(in_ + rr * hn);
            float ho = hf32[i];
            float hv = (1.f - zz) * nn + zz * ho;
            hf32[i] = hv;
            hf16[b * KP_ + j] = (_Float16)hv;
        }

        gbarrier(cnt, phase, &lphase);

        // logits = h_new · lin_W^T + lin_b : 64x128, WGs 0..3 (one C-tile/wave).
        if (wg < 4) {
            for (int i = tid * 8; i < 64 * KP_; i += NTHR_ * 8)
                async_copy_b128(ahB + i * 2, hf16, i * 2);
            wait_async0();
            __syncthreads();
            int tile = wg * 8 + wid;                 // 0..31
            int mt = tile >> 3, nt = tile & 7;       // 4 x 8 tiles
            v8f c = {};
#pragma unroll
            for (int k0 = 0; k0 < KP_; k0 += 32) {
                v16h a  = ld_frag(Ah + (mt * 16 + row) * KP_ + k0 + hih * 8);
                v16h bm = ld_frag(lin_h + (nt * 16 + row) * KP_ + k0 + hih * 8);
                c = __builtin_amdgcn_wmma_f32_16x16x32_f16(
                        false, a, false, bm, (short)0, c, false, false);
            }
            float lb = lin_b[nt * 16 + row];
#pragma unroll
            for (int r = 0; r < 8; ++r) {
                int m = mt * 16 + r + hih * 8;
                int n = nt * 16 + row;
                float v = c[r] + lb;
                logits[m * D_ + n] = v;
                if (s < L_ - 1) out_soft[(m * (L_ - 1) + s) * D_ + n] = v;
            }
        }

        gbarrier(cnt, phase, &lphase);

        // Greedy argmax feedback (first-max, matching jnp.argmax).
        if (wg == 0 && tid < B_) {
            const float* lr = logits + tid * D_;
            float best = lr[0];
            int bi = 0;
            for (int n2 = 1; n2 < D_; ++n2) {
                float v = lr[n2];
                if (v > best) { best = v; bi = n2; }
            }
            tok[tid] = bi;
            if (s < L_ - 1) out_asr[tid * (L_ - 1) + s] = (float)bi;
        }

        gbarrier(cnt, phase, &lphase);
    }
}

extern "C" void kernel_launch(void* const* d_in, const int* in_sizes, int n_in,
                              void* d_out, int out_size, void* d_ws, size_t ws_size,
                              hipStream_t stream)
{
    (void)in_sizes; (void)n_in; (void)out_size; (void)ws_size;

    const float* x      = (const float*)d_in[0];
    const int*   target = (const int*)  d_in[1];
    const float* emb    = (const float*)d_in[2];
    const float* eWih   = (const float*)d_in[3];
    const float* eWhh   = (const float*)d_in[4];
    const float* ebih   = (const float*)d_in[5];
    const float* ebhh   = (const float*)d_in[6];
    const float* dWih   = (const float*)d_in[7];
    const float* dWhh   = (const float*)d_in[8];
    const float* dbih   = (const float*)d_in[9];
    const float* dbhh   = (const float*)d_in[10];
    const float* linW   = (const float*)d_in[11];
    const float* linb   = (const float*)d_in[12];

    char* ws = (char*)d_ws;                       // ~3.4 MB used
    _Float16* eWih_h = (_Float16*)(ws + 0);       // 1024*320 f16
    _Float16* eWhh_h = (_Float16*)(ws + 655360);
    _Float16* dWih_h = (_Float16*)(ws + 1310720);
    _Float16* dWhh_h = (_Float16*)(ws + 1966080);
    _Float16* lin_h  = (_Float16*)(ws + 2621440); // 128*320 f16
    _Float16* emb_h  = (_Float16*)(ws + 2703360); // 128*320 f16
    float*    hf32   = (float*)   (ws + 2785280); // 64*304 f32
    _Float16* hf16   = (_Float16*)(ws + 2863104); // 64*320 f16
    float*    Gi     = (float*)   (ws + 2904064); // 64*1024 f32
    float*    Gh     = (float*)   (ws + 3166208); // 64*1024 f32
    float*    logits = (float*)   (ws + 3428352); // 64*128 f32
    int*      tok    = (int*)     (ws + 3461120); // 64 int
    int*      ctrl   = (int*)     (ws + 3461376); // barrier cnt/phase

    float* out_soft = (float*)d_out;                       // (B,127,D)
    float* out_tgt  = out_soft + B_ * (L_ - 1) * D_;       // (B*127,)
    float* out_asr  = out_tgt + B_ * (L_ - 1);             // (B,127,1)

    const size_t smem = (size_t)4 * 64 * KP_ * sizeof(_Float16);  // 160 KB / WGP
    hipFuncSetAttribute((const void*)enc_gru,
                        hipFuncAttributeMaxDynamicSharedMemorySize, (int)smem);
    hipFuncSetAttribute((const void*)dec_gru,
                        hipFuncAttributeMaxDynamicSharedMemorySize, (int)smem);

    // Reset state each call (graph-replay deterministic).
    hipMemsetAsync(hf32, 0, B_ * H_ * sizeof(float), stream);
    hipMemsetAsync(hf16, 0, B_ * KP_ * sizeof(_Float16), stream);
    hipMemsetAsync(ctrl, 0, 256, stream);

    pad_convert<<<512, 256, 0, stream>>>(eWih, eWih_h, GP_, G3_, F_);
    pad_convert<<<512, 256, 0, stream>>>(eWhh, eWhh_h, GP_, G3_, H_);
    pad_convert<<<512, 256, 0, stream>>>(dWih, dWih_h, GP_, G3_, F_);
    pad_convert<<<512, 256, 0, stream>>>(dWhh, dWhh_h, GP_, G3_, H_);
    pad_convert<<<512, 256, 0, stream>>>(linW, lin_h, D_, D_, H_);
    pad_convert<<<512, 256, 0, stream>>>(emb,  emb_h, D_, D_, F_);

    enc_gru<<<NWG_, NTHR_, smem, stream>>>(x, ebih, ebhh, eWih_h, eWhh_h,
                                           hf32, hf16, Gi, Gh, ctrl);
    dec_gru<<<NWG_, NTHR_, smem, stream>>>(target, emb_h, dbih, dbhh, dWih_h, dWhh_h,
                                           lin_h, linb, hf32, hf16, Gi, Gh,
                                           logits, tok, out_soft, out_asr, ctrl);
    copy_target<<<(B_ * (L_ - 1) + 255) / 256, 256, 0, stream>>>(target, out_tgt);
}
